// DVQVAE_Encoder_32598801776825
// MI455X (gfx1250) — matrix operand order repaired and checked by reading, not compiled, not run on hardware
//
#include <hip/hip_runtime.h>
#include <hip/hip_bf16.h>

// ---------------------------------------------------------------------------
// DVQVAE encoder for MI455X (gfx1250): all GEMMs via v_wmma_f32_16x16x32_f16,
// attention via flash-style streaming WMMA kernel. f32 in HBM, f16 in LDS/regs
// for matrix ops, f32 accumulation. GEMM uses 128x128 block tiles (8 waves,
// 32 FLOP/byte) to sit above the HBM roofline knee.
// ---------------------------------------------------------------------------

#define D_   512
#define NH_  8
#define DH_  64
#define FF_  2048
#define NL_  6
#define K_   1024
#define B_   8
#define T_   1024
#define SLD_ 512
#define ROWS_ (B_ * T_)   // 8192

typedef _Float16 h8  __attribute__((ext_vector_type(8)));
typedef _Float16 h16 __attribute__((ext_vector_type(16)));
typedef float    f8  __attribute__((ext_vector_type(8)));

__device__ __forceinline__ f8 wmma_f16(h16 a, h16 b, f8 c) {
    // D = A(16x32 f16) * B(32x16 f16) + C(16x16 f32)
    return __builtin_amdgcn_wmma_f32_16x16x32_f16(false, a, false, b,
                                                  (short)0, c, false, false);
}

// ---------------------------------------------------------------------------
// Tiled WMMA GEMM: C[M,N] = epilogue(A[M,K] @ W[N,K]^T + bias)
// Block = 256 threads (8 waves), tile 128(M) x 128(N), K-step 32.
// Wave (wm, wn) with wm = wave&3, wn = wave>>2 computes rows [wm*32,+32),
// cols [wn*64,+64): 2 resident A-frags x 4 B-frags = 8 WMMA per K-step.
// act bit0 = relu; residual added after act if non-null.
// Requires M%128==0, N%128==0, K%32==0 (true for all uses here).
// ---------------------------------------------------------------------------
__global__ __launch_bounds__(256) void gemm_kernel(
    const float* __restrict__ A, const float* __restrict__ W,
    const float* __restrict__ bias, const float* __restrict__ resid,
    float* __restrict__ C, int M, int N, int K, int act)
{
    __shared__ __align__(128) _Float16 Ash[128 * 32];
    __shared__ __align__(128) _Float16 Bsh[128 * 32];

    const int tid  = threadIdx.x;
    const int lane = tid & 31;
    const int wv   = tid >> 5;         // 0..7
    const int wm   = wv & 3;           // M sub-tile (32 rows)
    const int wn   = wv >> 2;          // N sub-tile (64 cols)
    const int g    = lane >> 4;        // lane group
    const int ln   = lane & 15;
    const int mBase = blockIdx.y * 128;
    const int nBase = blockIdx.x * 128;

    f8 acc[4][2] = {};                 // [n-tile][m-frag]

    const int ldRow = tid >> 1;          // 0..127
    const int ldK   = (tid & 1) * 16;    // 0 or 16

    for (int k0 = 0; k0 < K; k0 += 32) {
        const float* ap = A + (size_t)(mBase + ldRow) * K + k0 + ldK;
        const float* bp = W + (size_t)(nBase + ldRow) * K + k0 + ldK;
#pragma unroll
        for (int i = 0; i < 16; ++i) {
            Ash[ldRow * 32 + ldK + i] = (_Float16)ap[i];
            Bsh[ldRow * 32 + ldK + i] = (_Float16)bp[i];
        }
        if (k0 + 32 < K) {               // hint next K tile into cache
            __builtin_prefetch(ap + 32, 0, 0);
            __builtin_prefetch(bp + 32, 0, 0);
        }
        __syncthreads();

        // Two resident A fragments (rows wm*32+mf*16): lane holds row
        // (lane&15); halves: k = (h/8)*16 + g*8 + h%8
        h16 af[2];
#pragma unroll
        for (int mf = 0; mf < 2; ++mf) {
            const int row = wm * 32 + mf * 16 + ln;
            const h8* lo = (const h8*)&Ash[row * 32 + g * 8];
            const h8* hi = (const h8*)&Ash[row * 32 + 16 + g * 8];
            h8 l = *lo, h = *hi;
            af[mf] = __builtin_shufflevector(l, h, 0, 1, 2, 3, 4, 5, 6, 7,
                                                   8, 9, 10, 11, 12, 13, 14, 15);
        }
#pragma unroll
        for (int nt = 0; nt < 4; ++nt) {
            // B fragment: lane holds col (lane&15); halves k = (h/8)*16 + g*8 + h%8
            const h8* blo = (const h8*)&Bsh[(wn * 64 + nt * 16 + ln) * 32 + g * 8];
            const h8* bhi = (const h8*)&Bsh[(wn * 64 + nt * 16 + ln) * 32 + 16 + g * 8];
            h16 b = __builtin_shufflevector(*blo, *bhi, 0, 1, 2, 3, 4, 5, 6, 7,
                                            8, 9, 10, 11, 12, 13, 14, 15);
            acc[nt][0] = wmma_f16(af[0], b, acc[nt][0]);
            acc[nt][1] = wmma_f16(af[1], b, acc[nt][1]);
        }
        __syncthreads();
    }

#pragma unroll
    for (int nt = 0; nt < 4; ++nt)
#pragma unroll
        for (int mf = 0; mf < 2; ++mf)
#pragma unroll
            for (int r = 0; r < 8; ++r) {
                const int m = mBase + wm * 32 + mf * 16 + r + (g << 3);
                const int n = nBase + wn * 64 + nt * 16 + ln;
                float v = acc[nt][mf][r];
                if (bias)    v += bias[n];
                if (act & 1) v  = fmaxf(v, 0.0f);
                if (resid)   v += resid[(size_t)m * N + n];
                C[(size_t)m * N + n] = v;
            }
}

// ---------------------------------------------------------------------------
// Flash-style attention: one wave per (batch, head, 16-query tile).
// dh = 64, key tiles of 32. Scores/PV via WMMA, online softmax in regs.
// ---------------------------------------------------------------------------
__global__ __launch_bounds__(32) void attn_kernel(
    const float* __restrict__ qkv,   // [B, T, 3*D]
    float* __restrict__ out)         // [B, T, D]
{
    __shared__ __align__(128) _Float16 Psh[16 * 32];

    const int lane = threadIdx.x;
    const int g    = lane >> 4;
    const int ln   = lane & 15;
    const int bid  = blockIdx.x;
    const int qt   = bid & 63;         // T/16
    const int head = bid >> 6;
    const int b    = head >> 3;
    const int hh   = head & 7;
    const int QT   = qt * 16;
    const int RS   = 3 * D_;           // qkv row stride

    const float* base = qkv + (size_t)b * T_ * RS + hh * DH_;

    // Q fragments (pre-scaled by 1/sqrt(dh)=1/8), d = 0..31 and 32..63
    h16 qf[2];
#pragma unroll
    for (int f = 0; f < 2; ++f) {
        const float* qrow = base + (size_t)(QT + ln) * RS;
#pragma unroll
        for (int e = 0; e < 16; ++e) {
            const int dd = f * 32 + ((e >> 3) << 4) + g * 8 + (e & 7);
            qf[f][e] = (_Float16)(qrow[dd] * 0.125f);
        }
    }

    f8 acc[4] = {};
    float mrow[8], lrow[8];
#pragma unroll
    for (int r = 0; r < 8; ++r) { mrow[r] = -1e30f; lrow[r] = 0.0f; }

    for (int kt = 0; kt < T_ / 32; ++kt) {
        const int kb = kt * 32;
        f8 s0 = {}, s1 = {};

        // S = Q @ K^T  (two 16x16 tiles of a 16x32 score block)
#pragma unroll
        for (int st = 0; st < 2; ++st) {
            const float* krow = base + (size_t)(kb + st * 16 + ln) * RS + D_;
#pragma unroll
            for (int c = 0; c < 2; ++c) {
                h16 bf;
#pragma unroll
                for (int e = 0; e < 16; ++e)
                    bf[e] = (_Float16)krow[c * 32 + ((e >> 3) << 4) + g * 8 + (e & 7)];
                if (st == 0) s0 = wmma_f16(qf[c], bf, s0);
                else         s1 = wmma_f16(qf[c], bf, s1);
            }
        }

        // online softmax update, per output row r (row = r + 8*g)
#pragma unroll
        for (int r = 0; r < 8; ++r) {
            float mx = fmaxf(s0[r], s1[r]);
#pragma unroll
            for (int msk = 1; msk < 16; msk <<= 1)
                mx = fmaxf(mx, __shfl_xor(mx, msk, 32));
            const float mn = fmaxf(mrow[r], mx);
            const float sc = __expf(mrow[r] - mn);
            const float p0 = __expf(s0[r] - mn);
            const float p1 = __expf(s1[r] - mn);
            s0[r] = p0; s1[r] = p1;
            float rs = p0 + p1;
#pragma unroll
            for (int msk = 1; msk < 16; msk <<= 1)
                rs += __shfl_xor(rs, msk, 32);
            lrow[r] = lrow[r] * sc + rs;
            mrow[r] = mn;
#pragma unroll
            for (int c = 0; c < 4; ++c) acc[c][r] *= sc;
        }

        // P: C-layout -> LDS -> A-fragment layout
        __syncthreads();
#pragma unroll
        for (int r = 0; r < 8; ++r) {
            const int m = r + (g << 3);
            Psh[m * 32 + ln]      = (_Float16)s0[r];
            Psh[m * 32 + 16 + ln] = (_Float16)s1[r];
        }
        __syncthreads();
        h16 pf;
#pragma unroll
        for (int e = 0; e < 16; ++e) {
            const int kk = ((e >> 3) << 4) + g * 8 + (e & 7);
            pf[e] = Psh[ln * 32 + kk];
        }

        // acc += P @ V
#pragma unroll
        for (int c = 0; c < 4; ++c) {
            h16 vf;
#pragma unroll
            for (int e = 0; e < 16; ++e) {
                const int key = kb + ((e >> 3) << 4) + g * 8 + (e & 7);
                vf[e] = (_Float16)base[(size_t)key * RS + 2 * D_ + c * 16 + ln];
            }
            acc[c] = wmma_f16(pf, vf, acc[c]);
        }
    }

#pragma unroll
    for (int c = 0; c < 4; ++c)
#pragma unroll
        for (int r = 0; r < 8; ++r) {
            const int tq = QT + r + (g << 3);
            out[((size_t)b * T_ + tq) * D_ + hh * DH_ + c * 16 + ln] =
                acc[c][r] / lrow[r];
        }
}

// ---------------------------------------------------------------------------
// LayerNorm over D=512, optional residual add, relu (bit0), +PE (bit1).
// One 256-thread block per row.
// ---------------------------------------------------------------------------
__device__ __forceinline__ float pe_val(int t, int d) {
    const float de  = (float)(d & ~1);
    const float div = __expf(de * (-9.210340371976184f / (float)D_)); // ln(1e4)
    const float a   = (float)t * div;
    return (d & 1) ? __cosf(a) : __sinf(a);
}

__global__ __launch_bounds__(256) void ln_kernel(
    const float* __restrict__ X, const float* __restrict__ R,
    const float* __restrict__ gamma, const float* __restrict__ beta,
    float* __restrict__ out, int mode)
{
    __shared__ float red[256];
    const int row = blockIdx.x;
    const int tid = threadIdx.x;
    const float* xr = X + (size_t)row * D_;

    float x0 = xr[tid], x1 = xr[tid + 256];
    if (R) { x0 += R[(size_t)row * D_ + tid]; x1 += R[(size_t)row * D_ + tid + 256]; }

    red[tid] = x0 + x1; __syncthreads();
    for (int s = 128; s > 0; s >>= 1) { if (tid < s) red[tid] += red[tid + s]; __syncthreads(); }
    const float mean = red[0] * (1.0f / (float)D_);
    __syncthreads();
    const float d0 = x0 - mean, d1 = x1 - mean;
    red[tid] = d0 * d0 + d1 * d1; __syncthreads();
    for (int s = 128; s > 0; s >>= 1) { if (tid < s) red[tid] += red[tid + s]; __syncthreads(); }
    const float rstd = rsqrtf(red[0] * (1.0f / (float)D_) + 1e-5f);

    float y0 = d0 * rstd * gamma[tid]       + beta[tid];
    float y1 = d1 * rstd * gamma[tid + 256] + beta[tid + 256];
    if (mode & 1) { y0 = fmaxf(y0, 0.0f); y1 = fmaxf(y1, 0.0f); }
    if (mode & 2) {
        const int t = row & (T_ - 1);
        y0 += pe_val(t, tid);
        y1 += pe_val(t, tid + 256);
    }
    out[(size_t)row * D_ + tid]       = y0;
    out[(size_t)row * D_ + tid + 256] = y1;
}

// I_T = sigmoid(Hm . w3 + b3), one wave per row
__global__ __launch_bounds__(256) void rowdot_sigmoid_kernel(
    const float* __restrict__ Hm, const float* __restrict__ w3,
    const float* __restrict__ b3, float* __restrict__ I)
{
    const int lane = threadIdx.x & 31;
    const int row  = blockIdx.x * 8 + (threadIdx.x >> 5);
    const float* hr = Hm + (size_t)row * D_;
    float s = 0.0f;
    for (int i = lane; i < D_; i += 32) s += hr[i] * w3[i];
#pragma unroll
    for (int m = 16; m > 0; m >>= 1) s += __shfl_xor(s, m, 32);
    if (lane == 0) I[row] = 1.0f / (1.0f + __expf(-(s + b3[0])));
}

// codebook row squared norms, one wave per row
__global__ __launch_bounds__(256) void cnorm_kernel(
    const float* __restrict__ cb, float* __restrict__ cn)
{
    const int lane = threadIdx.x & 31;
    const int row  = blockIdx.x * 8 + (threadIdx.x >> 5);
    const float* cr = cb + (size_t)row * D_;
    float s = 0.0f;
    for (int i = lane; i < D_; i += 32) s += cr[i] * cr[i];
#pragma unroll
    for (int m = 16; m > 0; m >>= 1) s += __shfl_xor(s, m, 32);
    if (lane == 0) cn[row] = s;
}

// inclusive scan of I over T per batch, S = floor(cumsum), seg = int(S) clamped
__global__ __launch_bounds__(1024) void scan_kernel(
    const float* __restrict__ I, float* __restrict__ S, int* __restrict__ seg)
{
    __shared__ float sh[1024];
    const int b = blockIdx.x, tid = threadIdx.x;
    sh[tid] = I[b * T_ + tid]; __syncthreads();
    for (int off = 1; off < 1024; off <<= 1) {
        const float v = (tid >= off) ? sh[tid - off] : 0.0f;
        __syncthreads();
        sh[tid] += v;
        __syncthreads();
    }
    const float fl = floorf(sh[tid]);
    S[b * T_ + tid] = fl;
    int sg = (int)fl;
    if (sg < 0) sg = 0;
    if (sg > T_ - 1) sg = T_ - 1;
    seg[b * T_ + tid] = sg;
}

__global__ void zero_kernel(float* p, int n) {
    const int i = blockIdx.x * 256 + threadIdx.x;
    if (i < n) p[i] = 0.0f;
}

__global__ void copy_kernel(const float* __restrict__ s, float* __restrict__ d, int n) {
    const int i = blockIdx.x * 256 + threadIdx.x;
    if (i < n) d[i] = s[i];
}

// weighted segment scatter: Z[b, seg, :] += H[b,t,:] * I[b,t]; Dcnt[b,seg]++
__global__ __launch_bounds__(256) void scatter_kernel(
    const float* __restrict__ H, const float* __restrict__ I,
    const int* __restrict__ seg, float* __restrict__ Z, int* __restrict__ Dcnt)
{
    const int bt  = blockIdx.x;
    const int b   = bt >> 10;
    const int tid = threadIdx.x;
    const int sg  = seg[bt];
    const float w = I[bt];
    const float* hr = H + (size_t)bt * D_;
    float* zr = Z + ((size_t)(b * T_ + sg)) * D_;
    for (int d = tid; d < D_; d += 256) atomicAdd(&zr[d], hr[d] * w);
    if (tid == 0) atomicAdd(&Dcnt[b * T_ + sg], 1);
}

// argmin over (||c||^2 - 2 z.c) + gather z_q, one wave per row
__global__ __launch_bounds__(32) void argmin_kernel(
    const float* __restrict__ dot, const float* __restrict__ cn,
    const float* __restrict__ cb, int* __restrict__ idx, float* __restrict__ zq)
{
    const int row  = blockIdx.x;
    const int lane = threadIdx.x;
    float best = 3.4e38f; int bi = 0;
    for (int n = lane; n < K_; n += 32) {
        const float s = cn[n] - 2.0f * dot[(size_t)row * K_ + n];
        if (s < best) { best = s; bi = n; }
    }
#pragma unroll
    for (int m = 16; m > 0; m >>= 1) {
        const float ob = __shfl_xor(best, m, 32);
        const int   oi = __shfl_xor(bi, m, 32);
        if (ob < best || (ob == best && oi < bi)) { best = ob; bi = oi; }
    }
    bi = __shfl(bi, 0, 32);
    if (lane == 0) idx[row] = bi;
    const float* cr = cb + (size_t)bi * D_;
    float* zr = zq + (size_t)row * D_;
    for (int d = lane; d < D_; d += 32) zr[d] = cr[d];
}

// ---------------------------------------------------------------------------
extern "C" void kernel_launch(void* const* d_in, const int* in_sizes, int n_in,
                              void* d_out, int out_size, void* d_ws, size_t ws_size,
                              hipStream_t stream) {
    (void)in_sizes; (void)n_in; (void)out_size; (void)ws_size;

    const float* X_T   = (const float*)d_in[0];
    const float* W_emb = (const float*)d_in[1];
    const float* b_emb = (const float*)d_in[2];
    const float* ln_g  = (const float*)d_in[3];
    const float* ln_b  = (const float*)d_in[4];
    const float* qkv_w = (const float*)d_in[5];
    const float* qkv_b = (const float*)d_in[6];
    const float* out_w = (const float*)d_in[7];
    const float* out_b = (const float*)d_in[8];
    const float* ln1_g = (const float*)d_in[9];
    const float* ln1_b = (const float*)d_in[10];
    const float* ff1_w = (const float*)d_in[11];
    const float* ff1_b = (const float*)d_in[12];
    const float* ff2_w = (const float*)d_in[13];
    const float* ff2_b = (const float*)d_in[14];
    const float* ln2_g = (const float*)d_in[15];
    const float* ln2_b = (const float*)d_in[16];
    const float* W2_w  = (const float*)d_in[17];
    const float* W2_b  = (const float*)d_in[18];
    const float* W3_w  = (const float*)d_in[19];
    const float* W3_b  = (const float*)d_in[20];
    const float* codebook = (const float*)d_in[21];

    // workspace layout (bytes)
    const size_t MB = 1u << 20;
    char* ws = (char*)d_ws;
    float* buf_x   = (float*)(ws + 0 * MB);     // [8192, 512]  encoder state
    float* buf_a   = (float*)(ws + 16 * MB);    // [8192, 512]  attn / emb tmp
    float* buf_t   = (float*)(ws + 32 * MB);    // [8192, 512]  proj / ff / Hm tmp
    float* buf_big = (float*)(ws + 48 * MB);    // [8192, 2048] qkv / ffh / vq-dot
    float* cn      = (float*)(ws + 112 * MB);   // [1024] codebook norms
    int*   segb    = (int*)  (ws + 112 * MB + 4096); // [8192] segment ids

    // d_out layout (element offsets, return order)
    float* o_zq   = (float*)d_out;                 // [8192, 512]
    int*   o_Dcnt = (int*)  ((float*)d_out + 4194304); // [8192]
    float* o_S    = (float*)d_out + 4202496;       // [8192]
    float* o_Z    = (float*)d_out + 4210688;       // [8192, 512]
    float* o_I    = (float*)d_out + 8404992;       // [8192]
    int*   o_idx  = (int*)  ((float*)d_out + 8413184); // [8192]
    float* o_H    = (float*)d_out + 8421376;       // [8192, 512]

    const dim3 blk256(256), blk32(32), blk1024(1024);
    const int Mr = ROWS_;  // 8192 rows

    // 1. embedding GEMM: X_T @ W_emb^T + b_emb -> buf_a
    gemm_kernel<<<dim3(D_ / 128, Mr / 128), blk256, 0, stream>>>(
        X_T, W_emb, b_emb, nullptr, buf_a, Mr, D_, SLD_, 0);
    // 2. relu(LN(.)) + PE -> buf_x
    ln_kernel<<<Mr, blk256, 0, stream>>>(buf_a, nullptr, ln_g, ln_b, buf_x, 3);

    // 3. transformer layers
    for (int l = 0; l < NL_; ++l) {
        const float* qw = qkv_w + (size_t)l * 3 * D_ * D_;
        const float* qb = qkv_b + (size_t)l * 3 * D_;
        const float* ow = out_w + (size_t)l * D_ * D_;
        const float* ob = out_b + (size_t)l * D_;
        const float* f1w = ff1_w + (size_t)l * FF_ * D_;
        const float* f1b = ff1_b + (size_t)l * FF_;
        const float* f2w = ff2_w + (size_t)l * D_ * FF_;
        const float* f2b = ff2_b + (size_t)l * D_;

        gemm_kernel<<<dim3(3 * D_ / 128, Mr / 128), blk256, 0, stream>>>(
            buf_x, qw, qb, nullptr, buf_big, Mr, 3 * D_, D_, 0);
        attn_kernel<<<B_ * NH_ * (T_ / 16), blk32, 0, stream>>>(buf_big, buf_a);
        gemm_kernel<<<dim3(D_ / 128, Mr / 128), blk256, 0, stream>>>(
            buf_a, ow, ob, nullptr, buf_t, Mr, D_, D_, 0);
        ln_kernel<<<Mr, blk256, 0, stream>>>(
            buf_t, buf_x, ln1_g + l * D_, ln1_b + l * D_, buf_x, 0);

        gemm_kernel<<<dim3(FF_ / 128, Mr / 128), blk256, 0, stream>>>(
            buf_x, f1w, f1b, nullptr, buf_big, Mr, FF_, D_, 1 /*relu*/);
        gemm_kernel<<<dim3(D_ / 128, Mr / 128), blk256, 0, stream>>>(
            buf_big, f2w, f2b, nullptr, buf_t, Mr, D_, FF_, 0);
        ln_kernel<<<Mr, blk256, 0, stream>>>(
            buf_t, buf_x, ln2_g + l * D_, ln2_b + l * D_, buf_x, 0);
    }

    // 4. H_T output
    copy_kernel<<<(Mr * D_) / 256, blk256, 0, stream>>>(buf_x, o_H, Mr * D_);

    // 5. Hm = relu(H @ W2^T + b2) + H -> buf_t
    gemm_kernel<<<dim3(D_ / 128, Mr / 128), blk256, 0, stream>>>(
        buf_x, W2_w, W2_b, buf_x, buf_t, Mr, D_, D_, 1 /*relu then +resid*/);
    // 6. I_T = sigmoid(Hm . w3 + b3)
    rowdot_sigmoid_kernel<<<Mr / 8, blk256, 0, stream>>>(buf_t, W3_w, W3_b, o_I);
    // 7. cumsum / floor / seg
    scan_kernel<<<B_, blk1024, 0, stream>>>(o_I, o_S, segb);
    // 8. zero Z and Dcnt, then scatter weighted segment sums
    zero_kernel<<<(Mr * D_) / 256, blk256, 0, stream>>>(o_Z, Mr * D_);
    zero_kernel<<<Mr / 256, blk256, 0, stream>>>((float*)o_Dcnt, Mr);
    scatter_kernel<<<Mr, blk256, 0, stream>>>(buf_x, o_I, segb, o_Z, o_Dcnt);

    // 9. VQ: codebook norms, Z @ codebook^T via WMMA, argmin + gather
    cnorm_kernel<<<K_ / 8, blk256, 0, stream>>>(codebook, cn);
    gemm_kernel<<<dim3(K_ / 128, Mr / 128), blk256, 0, stream>>>(
        o_Z, codebook, nullptr, nullptr, buf_big, Mr, K_, D_, 0);
    argmin_kernel<<<Mr, blk32, 0, stream>>>(buf_big, cn, codebook, o_idx, o_zq);
}
